// HGTLinkPredictor_21131239096610
// MI455X (gfx1250) — matrix operand order repaired and checked by reading, not compiled, not run on hardware
//
#include <hip/hip_runtime.h>
#include <hip/hip_bf16.h>

// ---------------------------------------------------------------------------
// HGT link predictor for MI455X (gfx1250, wave32, WMMA).
// Dense GEMMs: v_wmma_f32_16x16x32_bf16, 128x64 block tile, 16x64 per wave,
// LDS-staged with async global->LDS loads (ASYNCcnt) when available.
// Relation transforms (a_rel/m_rel) are folded into the K/V projection
// weights so Q/K'/V' is a single N x 128 @ 128 x 384 GEMM.
// ---------------------------------------------------------------------------

#ifndef __has_builtin
#define __has_builtin(x) 0
#endif
#if __has_builtin(__builtin_amdgcn_global_load_async_to_lds_b128) && \
    __has_builtin(__builtin_amdgcn_s_wait_asynccnt)
#define USE_ASYNC_LDS 1
#else
#define USE_ASYNC_LDS 0
#endif

typedef __bf16 bf16x16 __attribute__((ext_vector_type(16)));
typedef __bf16 bf16x8  __attribute__((ext_vector_type(8)));
typedef float  f32x8   __attribute__((ext_vector_type(8)));
typedef int    v4i     __attribute__((vector_size(16)));

#define AS_GLOBAL __attribute__((address_space(1)))
#define AS_SHARED __attribute__((address_space(3)))

__device__ __forceinline__ unsigned short f2bf(float f) {
    unsigned u = __float_as_uint(f);
    u += 0x7FFFu + ((u >> 16) & 1u);           // round-to-nearest-even
    return (unsigned short)(u >> 16);
}
// Monotone float -> uint key (order preserving) for atomicMax-based segment max.
__device__ __forceinline__ unsigned fkey(float f) {
    unsigned u = __float_as_uint(f);
    return (u & 0x80000000u) ? ~u : (u | 0x80000000u);
}
__device__ __forceinline__ float funkey(unsigned k) {
    unsigned u = (k & 0x80000000u) ? (k & 0x7FFFFFFFu) : ~k;
    return __uint_as_float(u);
}

// Copy one 16B chunk global->LDS (async DMA path when the toolchain has it).
__device__ __forceinline__ void stage16(const uint4* g, uint4* l) {
#if USE_ASYNC_LDS
    __builtin_amdgcn_global_load_async_to_lds_b128(
        (AS_GLOBAL v4i*)g, (AS_SHARED v4i*)l, 0, 0);
#else
    *l = *g;
#endif
}

// ---------------------------------------------------------------------------
// Fold a_rel/m_rel into K/V projection weights; emit bf16 transposed weights
// Wt[384][128] (rows = output cols, order: Q | K' | V') and fp32 bias[384].
// grid = 384 blocks x 128 threads.
// ---------------------------------------------------------------------------
__global__ void prep_qkv_weights(const float* __restrict__ Wq, const float* __restrict__ bq,
                                 const float* __restrict__ Wk, const float* __restrict__ bk,
                                 const float* __restrict__ Wv, const float* __restrict__ bv,
                                 const float* __restrict__ a_rel, const float* __restrict__ m_rel,
                                 unsigned short* __restrict__ Wt, float* __restrict__ biasAll) {
    const int o = blockIdx.x;    // 0..383 output column
    const int c = threadIdx.x;   // 0..127 input row
    float w, b;
    if (o < 128) {
        w = Wq[(size_t)c * 128 + o];
        b = bq[o];
    } else {
        const int e = (o < 256) ? (o - 128) : (o - 256);
        const float* W    = (o < 256) ? Wk : Wv;
        const float* bsrc = (o < 256) ? bk : bv;
        const float* R    = (o < 256) ? a_rel : m_rel;   // [H][D][D]
        const int h = e >> 5, ec = e & 31;
        float s = 0.f, sb = 0.f;
#pragma unroll 8
        for (int d = 0; d < 32; ++d) {
            float rv = R[(size_t)h * 1024 + d * 32 + ec];
            s  += W[(size_t)c * 128 + h * 32 + d] * rv;
            sb += bsrc[h * 32 + d] * rv;
        }
        w = s; b = sb;
    }
    Wt[(size_t)o * 128 + c] = f2bf(w);
    if (c == 0) biasAll[o] = b;
}

// Transpose 128x128 fp32 weight -> bf16 [128][128] (rows = output cols).
__global__ void transpose_to_bf16(const float* __restrict__ W, unsigned short* __restrict__ Wt) {
    const int o = blockIdx.x, c = threadIdx.x;
    Wt[(size_t)o * 128 + c] = f2bf(W[(size_t)c * 128 + o]);
}

__global__ void f32_to_bf16(const float* __restrict__ in, unsigned short* __restrict__ out, size_t n) {
    size_t i = (size_t)blockIdx.x * blockDim.x + threadIdx.x;
    if (i < n) out[i] = f2bf(in[i]);
}

// ---------------------------------------------------------------------------
// C[M][ldc] = A[M][128](bf16) @ Bt[Ncols][128]^T (bf16, stored transposed)
//           + bias[Ncols]     (fp32 accumulate)
// Block: 256 threads = 8 waves; block tile 128 rows x 64 cols.
// Wave w owns rows [16w,16w+16) x all 64 cols: 4 accumulators, 16 WMMAs,
// A fragments reused 4x per k-step. grid = (ceil(M/128), Ncols/64).
// ---------------------------------------------------------------------------
__global__ __launch_bounds__(256) void gemm_bf16_wmma(
        const unsigned short* __restrict__ A, const unsigned short* __restrict__ Bt,
        const float* __restrict__ bias, float* __restrict__ C, int M, int ldc) {
    __shared__ unsigned short sA[128 * 128];  // 32 KB
    __shared__ unsigned short sB[64 * 128];   // 16 KB
    const int t = threadIdx.x;
    const int tile_m = blockIdx.x * 128;
    const int tile_n = blockIdx.y * 64;

    // ---- Stage A tile (128x128 bf16): 64 contiguous elems (128B) per thread.
    // OOB rows are clamped to M-1 (their outputs are masked at store time).
    {
        const int base = t * 64;
        const int row  = base >> 7, col = base & 127;
        int grow = tile_m + row;
        if (grow > M - 1) grow = M - 1;
        const uint4* s = (const uint4*)(A + (size_t)grow * 128 + col);
        uint4* d = (uint4*)(sA + base);
#pragma unroll
        for (int i = 0; i < 4; ++i) stage16(s + i, d + i);
    }
    // ---- Stage Bt tile (64x128 bf16): 32 contiguous elems (64B) per thread.
    {
        const int base = t * 32;
        const int row  = base >> 7, col = base & 127;
        const uint4* s = (const uint4*)(Bt + (size_t)(tile_n + row) * 128 + col);
        uint4* d = (uint4*)(sB + base);
#pragma unroll
        for (int i = 0; i < 4; ++i) stage16(s + i, d + i);
    }
    // ---- Second half of A staging.
    {
        const int base = t * 64;
        const int row  = base >> 7, col = (base & 127) + 32;
        int grow = tile_m + row;
        if (grow > M - 1) grow = M - 1;
        const uint4* s = (const uint4*)(A + (size_t)grow * 128 + col);
        uint4* d = (uint4*)(sA + base + 32);
#pragma unroll
        for (int i = 0; i < 4; ++i) stage16(s + i, d + i);
    }
#if USE_ASYNC_LDS
    __builtin_amdgcn_s_wait_asynccnt(0);
#endif
    __syncthreads();

    const int wave = t >> 5, lane = t & 31;
    const int hi16 = lane >> 4;                       // lane half
    const unsigned short* aRow = sA + (size_t)(wave * 16 + (lane & 15)) * 128;
    const unsigned short* bCol = sB + (size_t)(lane & 15) * 128;

    f32x8 acc[4] = {{}, {}, {}, {}};
#pragma unroll
    for (int s = 0; s < 4; ++s) {
        // A fragment (16x32): lane<16 -> K {0..7,16..23}, lane>=16 -> {8..15,24..31}
        const int kA = s * 32 + hi16 * 8;
        bf16x8 alo = *(const bf16x8*)(aRow + kA);
        bf16x8 ahi = *(const bf16x8*)(aRow + kA + 16);
        bf16x16 av;
#pragma unroll
        for (int i = 0; i < 8; ++i) { av[i] = alo[i]; av[i + 8] = ahi[i]; }
        // B fragments (32x16): lane<16 -> K {0..7,16..23}, lane>=16 -> {8..15,24..31}
        const int kB = s * 32 + hi16 * 8;
#pragma unroll
        for (int n = 0; n < 4; ++n) {
            const unsigned short* bRow = bCol + (size_t)n * 16 * 128;
            bf16x8 blo = *(const bf16x8*)(bRow + kB);
            bf16x8 bhi = *(const bf16x8*)(bRow + kB + 16);
            bf16x16 bv;
#pragma unroll
            for (int i = 0; i < 8; ++i) { bv[i] = blo[i]; bv[i + 8] = bhi[i]; }
            acc[n] = __builtin_amdgcn_wmma_f32_16x16x32_bf16(
                         false, av, false, bv, (short)0, acc[n], false, false);
        }
    }

    // D layout: VGPR r -> M = r + 8*(lane>=16), N = lane&15
    const int om = tile_m + wave * 16 + hi16 * 8;
    const bool full = (tile_m + 128) <= M;   // block-uniform fast path
#pragma unroll
    for (int n = 0; n < 4; ++n) {
        const int on = tile_n + n * 16 + (lane & 15);
        const float bb = bias[on];
        float* cp = C + (size_t)om * ldc + on;
        if (full) {
#pragma unroll
            for (int r = 0; r < 8; ++r) cp[(size_t)r * ldc] = acc[n][r] + bb;
        } else {
#pragma unroll
            for (int r = 0; r < 8; ++r)
                if (om + r < M) cp[(size_t)r * ldc] = acc[n][r] + bb;
        }
    }
}

// ---------------------------------------------------------------------------
// Edge pass 1: alpha[e,h] = dot(q[dst],k[src]) * p[h] / sqrt(D); segment max
// via atomicMax on order-preserving uint keys. One wave per edge.
// qkv layout per node: [q(0:128) | k(128:256) | v(256:384)]
// ---------------------------------------------------------------------------
__global__ void attn_logits(const float* __restrict__ qkv,
                            const int* __restrict__ src, const int* __restrict__ dst,
                            const float* __restrict__ p, float* __restrict__ alpha,
                            unsigned* __restrict__ amaxkey, int E, float inv_sqrt_d) {
    const int gw = (int)((blockIdx.x * blockDim.x + threadIdx.x) >> 5);
    const int lane = threadIdx.x & 31;
    if (gw >= E) return;
    const int s = src[gw], d = dst[gw];
    const float* q = qkv + (size_t)d * 384;
    const float* k = qkv + (size_t)s * 384 + 128;
#pragma unroll
    for (int h = 0; h < 4; ++h) {
        float part = q[h * 32 + lane] * k[h * 32 + lane];
#pragma unroll
        for (int off = 16; off > 0; off >>= 1) part += __shfl_down(part, off, 32);
        if (lane == 0) {
            const float al = part * p[h] * inv_sqrt_d;
            alpha[(size_t)gw * 4 + h] = al;
            atomicMax(&amaxkey[(size_t)d * 4 + h], fkey(al));
        }
    }
}

// Edge pass 2: ex = exp(alpha - max); denom[dst,h] += ex; agg[dst,:] += ex*v[src,:]
// (normalization ex/denom deferred to the node pass -- algebraically identical).
__global__ void attn_agg(const float* __restrict__ qkv,
                         const int* __restrict__ src, const int* __restrict__ dst,
                         const float* __restrict__ alpha, const unsigned* __restrict__ amaxkey,
                         float* __restrict__ denom, float* __restrict__ agg, int E) {
    const int gw = (int)((blockIdx.x * blockDim.x + threadIdx.x) >> 5);
    const int lane = threadIdx.x & 31;
    if (gw >= E) return;
    const int s = src[gw], d = dst[gw];
    const float* v = qkv + (size_t)s * 384 + 256;
    float* aggd = agg + (size_t)d * 128;
#pragma unroll
    for (int h = 0; h < 4; ++h) {
        const float am = funkey(amaxkey[(size_t)d * 4 + h]);
        const float ex = expf(alpha[(size_t)gw * 4 + h] - am);
        if (lane == 0) atomicAdd(&denom[(size_t)d * 4 + h], ex);
        atomicAdd(&aggd[h * 32 + lane], ex * v[h * 32 + lane]);
    }
}

// Node pass: normalize by denom, exact GELU, emit bf16 for the Wo GEMM.
__global__ void finalize_gelu(const float* __restrict__ agg, const float* __restrict__ denom,
                              unsigned short* __restrict__ gbf, size_t n) {
    size_t idx = (size_t)blockIdx.x * blockDim.x + threadIdx.x;
    if (idx >= n) return;
    const size_t i = idx >> 7;
    const int c = (int)(idx & 127), h = c >> 5;
    const float dn = denom[i * 4 + h];
    const float val = (dn > 0.f) ? agg[idx] / dn : 0.f;
    const float g = 0.5f * val * (1.f + erff(val * 0.70710678118654752f));
    gbf[idx] = f2bf(g);
}

// Trainable sigmoid skip: h = a*proj + (1-a)*x; also emit bf16 for next layer.
__global__ void skip_combine(const float* __restrict__ proj, const float* __restrict__ xin,
                             const float* __restrict__ sk, float* __restrict__ hout,
                             unsigned short* __restrict__ hbf, size_t n) {
    size_t i = (size_t)blockIdx.x * blockDim.x + threadIdx.x;
    if (i >= n) return;
    const float a = 1.f / (1.f + expf(-sk[0]));
    const float v = a * proj[i] + (1.f - a) * xin[i];
    hout[i] = v;
    hbf[i]  = f2bf(v);
}

// Link decoder: out[p] = [z[src]||z[dst]] . Wlp + blp
__global__ void decode_edges(const float* __restrict__ z, const int* __restrict__ ei,
                             const float* __restrict__ Wlp, const float* __restrict__ blp,
                             float* __restrict__ out, int P) {
    const int p = blockIdx.x * blockDim.x + threadIdx.x;
    if (p >= P) return;
    const int s = ei[p], d = ei[P + p];
    const float* zs = z + (size_t)s * 128;
    const float* zd = z + (size_t)d * 128;
    float acc = blp[0];
#pragma unroll 4
    for (int c = 0; c < 128; ++c) acc += zs[c] * Wlp[c] + zd[c] * Wlp[128 + c];
    out[p] = acc;
}

// ---------------------------------------------------------------------------
extern "C" void kernel_launch(void* const* d_in, const int* in_sizes, int n_in,
                              void* d_out, int out_size, void* d_ws, size_t ws_size,
                              hipStream_t stream) {
    (void)n_in; (void)out_size; (void)ws_size;
    const float* x   = (const float*)d_in[0];
    const int*   ei  = (const int*)d_in[1];       // [2][E]
    const int*   pos = (const int*)d_in[3];       // [2][P]
    const int*   neg = (const int*)d_in[4];
    const float* Wlp = (const float*)d_in[29];
    const float* blp = (const float*)d_in[30];

    const int N = in_sizes[0] / 128;
    const int E = in_sizes[1] / 2;
    const int P = in_sizes[3] / 2;

    // Workspace carve-up (256B aligned)
    char* wp = (char*)d_ws;
    auto alloc = [&](size_t bytes) -> void* {
        void* p = wp;
        wp += (bytes + 255) & ~(size_t)255;
        return p;
    };
    unsigned short* hbf = (unsigned short*)alloc((size_t)N * 128 * 2); // bf16 layer input
    float*    qkv       = (float*)alloc((size_t)N * 384 * 4);
    float*    alpha     = (float*)alloc((size_t)E * 4 * 4);
    unsigned* amax      = (unsigned*)alloc((size_t)N * 4 * 4);
    float*    denom     = (float*)alloc((size_t)N * 4 * 4);
    float*    agg       = (float*)alloc((size_t)N * 128 * 4);
    unsigned short* gbf = (unsigned short*)alloc((size_t)N * 128 * 2);
    float*    proj      = (float*)alloc((size_t)N * 128 * 4);
    float*    h1        = (float*)alloc((size_t)N * 128 * 4);
    float*    h2        = (float*)alloc((size_t)N * 128 * 4);
    unsigned short* Wt  = (unsigned short*)alloc((size_t)384 * 128 * 2);
    float*    biasAll   = (float*)alloc((size_t)384 * 4);
    unsigned short* WoT = (unsigned short*)alloc((size_t)128 * 128 * 2);

    const float inv_sqrt_d = 0.17677669529663687f;   // 1/sqrt(32)
    const size_t nelem = (size_t)N * 128;
    const unsigned nblk  = (unsigned)((nelem + 255) / 256);
    const unsigned eblks = (unsigned)((E + 7) / 8);  // 1 wave per edge, 8 edges/block
    dim3 blk(256);

    f32_to_bf16<<<dim3(nblk), blk, 0, stream>>>(x, hbf, nelem);

    const float* xin = x;
    for (int l = 0; l < 2; ++l) {
        const int base = 5 + l * 12;   // Wk,bk,Wq,bq,Wv,bv,a,m,p,Wo,bo,skip
        const float* Wk_ = (const float*)d_in[base + 0];
        const float* bk_ = (const float*)d_in[base + 1];
        const float* Wq_ = (const float*)d_in[base + 2];
        const float* bq_ = (const float*)d_in[base + 3];
        const float* Wv_ = (const float*)d_in[base + 4];
        const float* bv_ = (const float*)d_in[base + 5];
        const float* a_  = (const float*)d_in[base + 6];
        const float* m_  = (const float*)d_in[base + 7];
        const float* p_  = (const float*)d_in[base + 8];
        const float* Wo_ = (const float*)d_in[base + 9];
        const float* bo_ = (const float*)d_in[base + 10];
        const float* sk_ = (const float*)d_in[base + 11];

        prep_qkv_weights<<<dim3(384), dim3(128), 0, stream>>>(Wq_, bq_, Wk_, bk_, Wv_, bv_,
                                                              a_, m_, Wt, biasAll);
        transpose_to_bf16<<<dim3(128), dim3(128), 0, stream>>>(Wo_, WoT);

        // Fused Q|K'|V' projection: N x 128 @ 128 x 384 (WMMA bf16)
        gemm_bf16_wmma<<<dim3((N + 127) / 128, 6), blk, 0, stream>>>(hbf, Wt, biasAll, qkv, N, 384);

        (void)hipMemsetAsync(amax,  0, (size_t)N * 4 * 4,   stream);
        (void)hipMemsetAsync(denom, 0, (size_t)N * 4 * 4,   stream);
        (void)hipMemsetAsync(agg,   0, (size_t)N * 128 * 4, stream);

        attn_logits<<<dim3(eblks), blk, 0, stream>>>(qkv, ei, ei + E, p_, alpha, amax, E, inv_sqrt_d);
        attn_agg<<<dim3(eblks), blk, 0, stream>>>(qkv, ei, ei + E, alpha, amax, denom, agg, E);
        finalize_gelu<<<dim3(nblk), blk, 0, stream>>>(agg, denom, gbf, nelem);

        // Output projection: N x 128 @ 128 x 128 (WMMA bf16)
        gemm_bf16_wmma<<<dim3((N + 127) / 128, 2), blk, 0, stream>>>(gbf, WoT, bo_, proj, N, 128);

        float* hout = (l == 0) ? h1 : h2;
        skip_combine<<<dim3(nblk), blk, 0, stream>>>(proj, xin, sk_, hout, hbf, nelem);
        xin = hout;
    }

    const unsigned pblks = (unsigned)((P + 255) / 256);
    decode_edges<<<dim3(pblks), blk, 0, stream>>>(h2, pos, Wlp, blp, (float*)d_out, P);
    decode_edges<<<dim3(pblks), blk, 0, stream>>>(h2, neg, Wlp, blp, (float*)d_out + P, P);
}
